// SimpleAttention_25280177504460
// MI455X (gfx1250) — matrix operand order repaired and checked by reading, not compiled, not run on hardware
//
#include <hip/hip_runtime.h>
#include <hip/hip_bf16.h>
#include <stdint.h>

// ---------------------------------------------------------------------------
// Types for CDNA5 WMMA (wave32): v_wmma_f32_16x16x32_bf16
// ---------------------------------------------------------------------------
typedef __bf16 bf16_t;
typedef __attribute__((ext_vector_type(16))) __bf16 v16bf;
typedef __attribute__((ext_vector_type(8)))  __bf16 v8bf;
typedef __attribute__((ext_vector_type(8)))  float  v8f;
typedef int v4i_vs __attribute__((vector_size(16)));   // matches builtin param type

union frag16 { v16bf v; v8bf h[2]; };

__device__ __forceinline__ bf16_t f2bf(float f) {
    unsigned u = __builtin_bit_cast(unsigned, f);
    u += 0x7fffu + ((u >> 16) & 1u);           // round-to-nearest-even
    unsigned short s = (unsigned short)(u >> 16);
    return __builtin_bit_cast(bf16_t, s);
}

__device__ __forceinline__ v8f wmma_bf16(v16bf a, v16bf b, v8f c) {
    return __builtin_amdgcn_wmma_f32_16x16x32_bf16(
        /*neg_a=*/false, a, /*neg_b=*/false, b,
        /*c_mod=*/(short)0, c, /*reuse_a=*/false, /*reuse_b=*/false);
}

// ---------------------------------------------------------------------------
// Async global->LDS copy (CDNA5 GLOBAL_LOAD_ASYNC_TO_LDS_B128, ASYNCcnt).
// Guarded: falls back to a synchronous 16B copy if the builtin is absent.
// ---------------------------------------------------------------------------
#if defined(__has_builtin)
#  if __has_builtin(__builtin_amdgcn_global_load_async_to_lds_b128)
#    define USE_ASYNC_LDS 1
#  endif
#endif
#ifndef USE_ASYNC_LDS
#  define USE_ASYNC_LDS 0
#endif

__device__ __forceinline__ void cp16_async(bf16_t* lds, const bf16_t* g) {
#if USE_ASYNC_LDS
    __builtin_amdgcn_global_load_async_to_lds_b128(
        (__attribute__((address_space(1))) v4i_vs*)(g),
        (__attribute__((address_space(3))) v4i_vs*)(lds),
        /*offset=*/0, /*cpol=*/0);
#else
    *(v8bf*)lds = *(const v8bf*)g;
#endif
}

#if USE_ASYNC_LDS
#  define WAIT_ASYNC(n) asm volatile("s_wait_asynccnt %0" :: "n"(n) : "memory")
#else
#  define WAIT_ASYNC(n) do {} while (0)
#endif

// ---------------------------------------------------------------------------
// fp32 -> bf16 conversion prepass (trivially small vs compute)
// ---------------------------------------------------------------------------
__global__ __launch_bounds__(256) void cvt_f32_bf16(const float* __restrict__ src,
                                                    bf16_t* __restrict__ dst, int n) {
    int i = blockIdx.x * blockDim.x + threadIdx.x;
    int stride = gridDim.x * blockDim.x;
    for (; i < n; i += stride) dst[i] = f2bf(src[i]);
}

// ---------------------------------------------------------------------------
// GEMM: C[M,N] = alpha * (A[M,K] @ W[N,K]^T + bias[N]); A,W bf16 K-contiguous.
// Double-buffered LDS tiles fed by async-to-LDS; 8 waves, each 32x64 (2x4 acc).
// ---------------------------------------------------------------------------
#define G_BM 128
#define G_BN 128
#define G_BK 32
#define G_LDA 40   // 32 + 8 halves pad -> 80B rows (16B aligned)
#define G_LDB 40

template <bool F32OUT>
__global__ __launch_bounds__(256)
void gemm_bf16_wmma(const bf16_t* __restrict__ A, const bf16_t* __restrict__ W,
                    const float* __restrict__ bias, float alpha,
                    bf16_t* __restrict__ Cb, float* __restrict__ Cf,
                    int M, int N, int K) {
    __shared__ bf16_t As[2][G_BM * G_LDA];
    __shared__ bf16_t Ws[2][G_BN * G_LDB];

    const int tid  = threadIdx.x;
    const int wave = tid >> 5;
    const int lane = tid & 31;
    const int fr   = lane & 15;    // frag row (A) / col (B,C)
    const int fh   = lane >> 4;    // half-wave select

    const int m0 = blockIdx.y * G_BM;
    const int n0 = blockIdx.x * G_BN;
    const int wm = wave >> 1;      // 0..3  -> 32-row strip
    const int wn = wave & 1;       // 0..1  -> 64-col strip

    const int srow = tid >> 1;     // staging row 0..127
    const int sseg = tid & 1;      // 16-half segment

    v8f acc[2][4];
#pragma unroll
    for (int i = 0; i < 2; ++i)
#pragma unroll
        for (int j = 0; j < 4; ++j) acc[i][j] = (v8f){0.f,0.f,0.f,0.f,0.f,0.f,0.f,0.f};

    // 4 async 16B transfers per thread per tile
    auto stageAW = [&](int buf, int k0) {
        const bf16_t* ga = A + (size_t)(m0 + srow) * K + k0 + sseg * 16;
        bf16_t* la = &As[buf][srow * G_LDA + sseg * 16];
        cp16_async(la, ga);
        cp16_async(la + 8, ga + 8);
        const bf16_t* gw = W + (size_t)(n0 + srow) * K + k0 + sseg * 16;
        bf16_t* lw = &Ws[buf][srow * G_LDB + sseg * 16];
        cp16_async(lw, gw);
        cp16_async(lw + 8, gw + 8);
    };

    const int NT = K / G_BK;
    stageAW(0, 0);
    for (int kb = 0; kb < NT; ++kb) {
        const int cur = kb & 1;
        if (kb + 1 < NT) {
            stageAW(cur ^ 1, (kb + 1) * G_BK);   // prefetch next tile (async)
            WAIT_ASYNC(4);                       // current tile's 4 have landed
        } else {
            WAIT_ASYNC(0);
        }
        __syncthreads();

        v16bf af[2], bfm[4];
#pragma unroll
        for (int i = 0; i < 2; ++i) {
            const bf16_t* p = &As[cur][(wm * 32 + i * 16 + fr) * G_LDA + fh * 8];
            frag16 u;
            u.h[0] = *(const v8bf*)(p);        // K = base+0..7
            u.h[1] = *(const v8bf*)(p + 16);   // K = base+16..23
            af[i] = u.v;
        }
#pragma unroll
        for (int j = 0; j < 4; ++j) {
            const bf16_t* p = &Ws[cur][(wn * 64 + j * 16 + fr) * G_LDB + fh * 8];
            frag16 u;
            u.h[0] = *(const v8bf*)(p);        // K = base+0..7
            u.h[1] = *(const v8bf*)(p + 16);   // K = base+16..23
            bfm[j] = u.v;
        }
#pragma unroll
        for (int i = 0; i < 2; ++i)
#pragma unroll
            for (int j = 0; j < 4; ++j)
                acc[i][j] = wmma_bf16(af[i], bfm[j], acc[i][j]);
        __syncthreads();
    }

    // ---- epilogue: C layout = (col = fr, rows = fh*8 + r), no runtime branch
#pragma unroll
    for (int i = 0; i < 2; ++i) {
#pragma unroll
        for (int j = 0; j < 4; ++j) {
            const int col = n0 + wn * 64 + j * 16 + fr;
            const float bv = bias[col];
#pragma unroll
            for (int r = 0; r < 8; ++r) {
                const int row = m0 + wm * 32 + i * 16 + fh * 8 + r;
                const float v = alpha * (acc[i][j][r] + bv);
                if constexpr (F32OUT) Cf[(size_t)row * N + col] = v;
                else                  Cb[(size_t)row * N + col] = f2bf(v);
            }
        }
    }
}

// ---------------------------------------------------------------------------
// Flash attention per (q-block of 128, head). D = 64, S = 4096, E = 1024.
// Q pre-scaled by 1/sqrt(D). One wave owns 16 query rows.
// K tile async-to-LDS double buffered; V staged transposed (sync scatter).
// ---------------------------------------------------------------------------
#define A_BQ  128
#define A_BKV 128
#define A_D   64
#define A_LDK 72    // K tile row stride (halves): 144B rows
#define A_LDV 136   // Vt row stride (halves): 272B rows
#define A_LDP 136   // P scratch row stride

__global__ __launch_bounds__(256)
void attn_flash_wmma(const bf16_t* __restrict__ Qb, const bf16_t* __restrict__ Kb,
                     const bf16_t* __restrict__ Vb, bf16_t* __restrict__ Ob,
                     int S, int E) {
    __shared__ bf16_t Ks[2][A_BKV * A_LDK];     // 2 x 18 KB  ([key][d])
    __shared__ bf16_t Vt[2][A_D * A_LDV];       // 2 x 17 KB  ([d][key], transposed)
    __shared__ bf16_t Ps[8 * 16 * A_LDP];       // 34 KB per-wave P scratch

    const int tid  = threadIdx.x;
    const int wave = tid >> 5;
    const int lane = tid & 31;
    const int fr   = lane & 15;
    const int fh   = lane >> 4;

    const int h  = blockIdx.y;
    const int q0 = blockIdx.x * A_BQ;
    const size_t hoff = (size_t)h * A_D;

    // ---- Q fragments (A-layout), rows q0 + wave*16 + fr, held for whole kernel
    const bf16_t* qrow = Qb + (size_t)(q0 + wave * 16 + fr) * E + hoff;
    v16bf qa[2];
#pragma unroll
    for (int t = 0; t < 2; ++t) {
        frag16 u;
        u.h[0] = *(const v8bf*)(qrow + t * 32 + fh * 8);
        u.h[1] = *(const v8bf*)(qrow + t * 32 + fh * 8 + 16);
        qa[t] = u.v;
    }

    float m_run[8], l_run[8];
#pragma unroll
    for (int r = 0; r < 8; ++r) { m_run[r] = -3.0e38f; l_run[r] = 0.f; }
    v8f o[4];
#pragma unroll
    for (int j = 0; j < 4; ++j) o[j] = (v8f){0.f,0.f,0.f,0.f,0.f,0.f,0.f,0.f};

    bf16_t* Pw = &Ps[wave * 16 * A_LDP];
    const int srow = tid >> 1;   // 0..127
    const int sseg = tid & 1;    // 32-half segment of the 64-wide row

    // K tile: 4 async 16B per thread; V tile: sync vector load + LDS transpose
    auto stageKV = [&](int buf, int k0) {
        const bf16_t* gk = Kb + (size_t)(k0 + srow) * E + hoff + sseg * 32;
        bf16_t* lk = &Ks[buf][srow * A_LDK + sseg * 32];
        cp16_async(lk,      gk);
        cp16_async(lk + 8,  gk + 8);
        cp16_async(lk + 16, gk + 16);
        cp16_async(lk + 24, gk + 24);
        v8bf vr[4];
        const v8bf* sv = (const v8bf*)(Vb + (size_t)(k0 + srow) * E + hoff + sseg * 32);
        vr[0] = sv[0]; vr[1] = sv[1]; vr[2] = sv[2]; vr[3] = sv[3];
#pragma unroll
        for (int g = 0; g < 4; ++g)
#pragma unroll
            for (int i = 0; i < 8; ++i)
                Vt[buf][(sseg * 32 + g * 8 + i) * A_LDV + srow] = vr[g][i];
    };

    const int NT = S / A_BKV;
    stageKV(0, 0);
    for (int kb = 0; kb < NT; ++kb) {
        const int cur = kb & 1;
        if (kb + 1 < NT) {
            stageKV(cur ^ 1, (kb + 1) * A_BKV);  // prefetch next K/V tile
            WAIT_ASYNC(4);
        } else {
            WAIT_ASYNC(0);
        }
        __syncthreads();

        // ---- scores: 8 key chunks, contract D=64 in 2 WMMA steps each ----
        v8f sfr[8];
#pragma unroll
        for (int c = 0; c < 8; ++c) {
            const bf16_t* kp = &Ks[cur][(c * 16 + fr) * A_LDK + fh * 8];
            frag16 b0, b1;
            b0.h[0] = *(const v8bf*)(kp);           // D 0..7 / 16..23 (+8 for fh)
            b0.h[1] = *(const v8bf*)(kp + 16);
            b1.h[0] = *(const v8bf*)(kp + 32);      // D 32..39 / 48..55 (+8 for fh)
            b1.h[1] = *(const v8bf*)(kp + 48);
            v8f s = (v8f){0.f,0.f,0.f,0.f,0.f,0.f,0.f,0.f};
            s = wmma_bf16(qa[0], b0.v, s);
            s = wmma_bf16(qa[1], b1.v, s);
            sfr[c] = s;
        }

        // ---- online softmax (rows = fh*8 + r; cols spread over 16 lanes) ----
        float mnew[8], alf[8];
#pragma unroll
        for (int r = 0; r < 8; ++r) {
            float mx = m_run[r];
#pragma unroll
            for (int c = 0; c < 8; ++c) mx = fmaxf(mx, sfr[c][r]);
#pragma unroll
            for (int off = 1; off < 16; off <<= 1)
                mx = fmaxf(mx, __shfl_xor(mx, off, 32));
            mnew[r] = mx;
            alf[r]  = __expf(m_run[r] - mx);
            m_run[r] = mx;
            l_run[r] *= alf[r];
        }
#pragma unroll
        for (int c = 0; c < 8; ++c) {
#pragma unroll
            for (int r = 0; r < 8; ++r) {
                float p = __expf(sfr[c][r] - mnew[r]);
                sfr[c][r] = p;
                Pw[(fh * 8 + r) * A_LDP + c * 16 + fr] = f2bf(p);
            }
        }
#pragma unroll
        for (int r = 0; r < 8; ++r) {
            float ps = 0.f;
#pragma unroll
            for (int c = 0; c < 8; ++c) ps += sfr[c][r];
#pragma unroll
            for (int off = 1; off < 16; off <<= 1)
                ps += __shfl_xor(ps, off, 32);
            l_run[r] += ps;
        }
#pragma unroll
        for (int j = 0; j < 4; ++j)
#pragma unroll
            for (int r = 0; r < 8; ++r) o[j][r] *= alf[r];

        // P scratch writes must be visible to this wave's ds loads
        asm volatile("s_wait_dscnt 0" ::: "memory");

        // ---- O += P @ V : contract 128 keys in 4 steps of 32 ----
#pragma unroll
        for (int j = 0; j < 4; ++j) {
#pragma unroll
            for (int t = 0; t < 4; ++t) {
                const bf16_t* pp = &Pw[fr * A_LDP + t * 32 + fh * 8];
                frag16 pa;
                pa.h[0] = *(const v8bf*)(pp);
                pa.h[1] = *(const v8bf*)(pp + 16);
                const bf16_t* vp = &Vt[cur][(j * 16 + fr) * A_LDV + t * 32 + fh * 8];
                frag16 vb;
                vb.h[0] = *(const v8bf*)(vp);
                vb.h[1] = *(const v8bf*)(vp + 16);
                o[j] = wmma_bf16(pa.v, vb.v, o[j]);
            }
        }
        __syncthreads();
    }

    // ---- finalize & store bf16 to attn buffer [S, E] ----
#pragma unroll
    for (int r = 0; r < 8; ++r) l_run[r] = 1.0f / l_run[r];
#pragma unroll
    for (int j = 0; j < 4; ++j) {
#pragma unroll
        for (int r = 0; r < 8; ++r) {
            const int row = q0 + wave * 16 + fh * 8 + r;
            Ob[(size_t)row * E + hoff + j * 16 + fr] = f2bf(o[j][r] * l_run[r]);
        }
    }
}

// ---------------------------------------------------------------------------
// Host-side launch
// ---------------------------------------------------------------------------
extern "C" void kernel_launch(void* const* d_in, const int* in_sizes, int n_in,
                              void* d_out, int out_size, void* d_ws, size_t ws_size,
                              hipStream_t stream) {
    (void)in_sizes; (void)n_in; (void)out_size; (void)ws_size;
    const int S = 4096, E = 1024;
    const float scale = 0.125f;  // 1/sqrt(64)

    const float* x  = (const float*)d_in[0];
    const float* Wq = (const float*)d_in[1];
    const float* bq = (const float*)d_in[2];
    const float* Wk = (const float*)d_in[3];
    const float* bk = (const float*)d_in[4];
    const float* Wv = (const float*)d_in[5];
    const float* bv = (const float*)d_in[6];
    const float* Wo = (const float*)d_in[7];
    const float* bo = (const float*)d_in[8];
    float* out = (float*)d_out;

    char* ws = (char*)d_ws;
    const size_t MB = 1u << 20;
    bf16_t* xb   = (bf16_t*)(ws + 0 * MB);   // 8 MB
    bf16_t* wqb  = (bf16_t*)(ws + 8 * MB);   // 2 MB
    bf16_t* wkb  = (bf16_t*)(ws + 10 * MB);  // 2 MB
    bf16_t* wvb  = (bf16_t*)(ws + 12 * MB);  // 2 MB
    bf16_t* wob  = (bf16_t*)(ws + 14 * MB);  // 2 MB
    bf16_t* Qb   = (bf16_t*)(ws + 16 * MB);  // 8 MB
    bf16_t* Kb   = (bf16_t*)(ws + 24 * MB);  // 8 MB
    bf16_t* Vb   = (bf16_t*)(ws + 32 * MB);  // 8 MB
    bf16_t* Ab   = (bf16_t*)(ws + 40 * MB);  // 8 MB (attention output)

    cvt_f32_bf16<<<2048, 256, 0, stream>>>(x,  xb,  S * E);
    cvt_f32_bf16<<<1024, 256, 0, stream>>>(Wq, wqb, E * E);
    cvt_f32_bf16<<<1024, 256, 0, stream>>>(Wk, wkb, E * E);
    cvt_f32_bf16<<<1024, 256, 0, stream>>>(Wv, wvb, E * E);
    cvt_f32_bf16<<<1024, 256, 0, stream>>>(Wo, wob, E * E);

    dim3 gg(E / G_BN, S / G_BM);   // (8, 32)
    gemm_bf16_wmma<false><<<gg, 256, 0, stream>>>(xb, wqb, bq, scale, Qb, nullptr, S, E, E);
    gemm_bf16_wmma<false><<<gg, 256, 0, stream>>>(xb, wkb, bk, 1.0f,  Kb, nullptr, S, E, E);
    gemm_bf16_wmma<false><<<gg, 256, 0, stream>>>(xb, wvb, bv, 1.0f,  Vb, nullptr, S, E, E);

    attn_flash_wmma<<<dim3(S / A_BQ, 16), 256, 0, stream>>>(Qb, Kb, Vb, Ab, S, E);

    gemm_bf16_wmma<true><<<gg, 256, 0, stream>>>(Ab, wob, bo, 1.0f, nullptr, out, S, E, E);
}
